// CabiNetCollision_8323646619967
// MI455X (gfx1250) — compile-verified, matrix operand order checked by reading, not run on hardware
//
#include <hip/hip_runtime.h>
#include <hip/hip_bf16.h>
#include <stdint.h>

// ---------------------------------------------------------------------------
// CabiNet collision net on MI455X (gfx1250, wave32, WMMA).
// Heavy GEMMs (conv3d implicit-GEMM, cls0, cls1) -> v_wmma_f32_16x16x32_bf16,
// 128x128x32 block tiles, 32x64 wave tiles, double-buffered LDS with
// register-staged global prefetch so HBM latency hides behind WMMA.
// Irregular graph ops (FPS, ball query, scatter-max, tiny MLPs) -> scalar.
// ---------------------------------------------------------------------------

typedef __bf16 bh;
typedef __attribute__((ext_vector_type(16))) __bf16 v16bf;
typedef __attribute__((ext_vector_type(8)))  float  v8f;

namespace {
constexpr int   kB      = 2;
constexpr int   kNScene = 32768;
constexpr int   kNObj   = 1024;
constexpr int   kQ      = 2048;
constexpr float kVox    = 0.0625f;
constexpr float kLow    = -0.5f;

constexpr int BM = 128;   // GEMM block tile M
constexpr int BN = 128;   // GEMM block tile N
constexpr int BK = 32;    // GEMM block tile K (== WMMA K for bf16)
}

// Load one 16x32(A) / 32x16(B) bf16 WMMA fragment from an LDS row of 32 halfs.
// Lane l (wave32): row/col = l&15, K-base kb = (l>>4)*8.
// v16bf elements 0..7  <- K = kb..kb+7 ; elements 8..15 <- K = kb+16..kb+23
__device__ __forceinline__ v16bf frag_from_lds(const bh* row, int kb) {
  union { uint4 u[2]; v16bf v; } c;
  c.u[0] = *(const uint4*)(row + kb);
  c.u[1] = *(const uint4*)(row + kb + 16);
  return c.v;
}

// ---------------------------------------------------------------------------
// Generic bf16 GEMM: C[M,N] = relu(A[M,K] * W[N,K]^T + bias)
// Requires M%128==0, N%128==0, K%32==0 (caller pads).
// 256 threads = 8 waves; wave (wm,wn) computes a 32x64 tile as 2x4 WMMAs.
// ---------------------------------------------------------------------------
__global__ __launch_bounds__(256)
void gemm_bf16_relu(const bh* __restrict__ A, const bh* __restrict__ W,
                    const float* __restrict__ bias, float* __restrict__ C,
                    int M, int N, int K, int relu) {
  __shared__ bh sA[2][BM][BK];   // 16 KB
  __shared__ bh sW[2][BN][BK];   // 16 KB
  const int tid  = threadIdx.x;
  const int lane = tid & 31;
  const int wave = tid >> 5;
  const int wm   = (wave & 3) * 32;   // 4 waves cover 128 rows
  const int wn   = (wave >> 2) * 64;  // 2 waves cover 128 cols
  const int m0   = blockIdx.x * BM;
  const int n0   = blockIdx.y * BN;
  const int r    = lane & 15;
  const int kb   = (lane >> 4) << 3;

  // Register staging: 512 16B-vectors per tile / 256 threads = 2 each.
  uint4 ra[2], rw[2];
  auto preload = [&](int k0) {
#pragma unroll
    for (int i = 0; i < 2; ++i) {
      int v = tid + i * 256;
      int row = v >> 2, kk = (v & 3) << 3;  // 4 vecs per 32-half row
      ra[i] = *(const uint4*)&A[(size_t)(m0 + row) * K + k0 + kk];
      rw[i] = *(const uint4*)&W[(size_t)(n0 + row) * K + k0 + kk];
    }
  };
  auto stage = [&](int buf) {
#pragma unroll
    for (int i = 0; i < 2; ++i) {
      int v = tid + i * 256;
      int row = v >> 2, kk = (v & 3) << 3;
      *(uint4*)&sA[buf][row][kk] = ra[i];
      *(uint4*)&sW[buf][row][kk] = rw[i];
    }
  };

  v8f acc[2][4] = {};
  preload(0);
  int buf = 0;
  for (int k0 = 0; k0 < K; k0 += BK) {
    stage(buf);
    __syncthreads();
    if (k0 + BK < K) preload(k0 + BK);  // overlaps with WMMA below

    v16bf af[2], bf[4];
#pragma unroll
    for (int i = 0; i < 2; ++i)
      af[i] = frag_from_lds(&sA[buf][wm + i * 16 + r][0], kb);
#pragma unroll
    for (int j = 0; j < 4; ++j)
      bf[j] = frag_from_lds(&sW[buf][wn + j * 16 + r][0], kb);
#pragma unroll
    for (int i = 0; i < 2; ++i)
#pragma unroll
      for (int j = 0; j < 4; ++j)
        acc[i][j] = __builtin_amdgcn_wmma_f32_16x16x32_bf16(
            false, af[i], false, bf[j], (short)0, acc[i][j], false, false);
    buf ^= 1;
    // single barrier per step is safe with double buffering (next stage
    // writes the other buffer; buffer reuse is fenced by the next barrier)
  }

  const int rhalf = (lane >> 4) << 3;  // C layout: lane>=16 -> M += 8
  for (int ti = 0; ti < 2; ++ti)
    for (int tj = 0; tj < 4; ++tj)
#pragma unroll
      for (int i = 0; i < 8; ++i) {
        int m = m0 + wm + ti * 16 + i + rhalf;
        int n = n0 + wn + tj * 16 + (lane & 15);
        if (m < M && n < N) {
          float v = acc[ti][tj][i] + bias[n];
          if (relu) v = fmaxf(v, 0.f);
          C[(size_t)m * N + n] = v;
        }
      }
}

// ---------------------------------------------------------------------------
// 3D conv (3x3x3, pad 1) as implicit GEMM with WMMA.
// in : [Cin][D^3] fp32 (per batch via blockIdx.z), gathered+cvt to bf16 in LDS
// W  : [OC][Cin*27] bf16,  out: [OC][D^3] fp32, relu(x + bias).
// M = D^3 (voxels), N = OC, K = Cin*27. Same pipelined 128x128x32 scheme.
// ---------------------------------------------------------------------------
template <int D>
__global__ __launch_bounds__(256)
void conv3d_wmma(const float* __restrict__ in, const bh* __restrict__ W,
                 const float* __restrict__ bias, float* __restrict__ out,
                 int Cin, int OC) {
  constexpr int M = D * D * D;
  const int K = Cin * 27;
  in  += (size_t)blockIdx.z * Cin * M;
  out += (size_t)blockIdx.z * OC * M;

  __shared__ bh sA[2][BM][BK];
  __shared__ bh sW[2][BN][BK];
  const int tid  = threadIdx.x;
  const int lane = tid & 31;
  const int wave = tid >> 5;
  const int wm   = (wave & 3) * 32;
  const int wn   = (wave >> 2) * 64;
  const int m0   = blockIdx.x * BM;
  const int n0   = blockIdx.y * BN;
  const int r    = lane & 15;
  const int kb   = (lane >> 4) << 3;

  float ga[16];    // gathered A elements (BM*BK / 256 threads)
  uint4 rw[2];
  auto preload = [&](int k0) {
#pragma unroll
    for (int i = 0; i < 16; ++i) {
      int idx = tid + i * 256;       // 0..4095
      int ml = idx >> 5, kk = idx & 31;
      int k  = k0 + kk;
      int ci = k / 27;
      int rem = k - ci * 27;
      int kd = rem / 9;  rem -= kd * 9;
      int kh = rem / 3;
      int kw = rem - kh * 3;
      int m = m0 + ml;
      int d0, d1, d2;
      if (D == 16) { d0 = m >> 8; d1 = (m >> 4) & 15; d2 = m & 15; }
      else         { d0 = m >> 6; d1 = (m >> 3) & 7;  d2 = m & 7;  }
      int z0 = d0 + kd - 1, z1 = d1 + kh - 1, z2 = d2 + kw - 1;
      float v = 0.f;
      if ((unsigned)z0 < (unsigned)D && (unsigned)z1 < (unsigned)D &&
          (unsigned)z2 < (unsigned)D)
        v = in[(size_t)ci * M + (z0 * D + z1) * D + z2];
      ga[i] = v;
    }
#pragma unroll
    for (int i = 0; i < 2; ++i) {
      int v = tid + i * 256;
      int nn = v >> 2, kk = (v & 3) << 3;
      rw[i] = *(const uint4*)&W[(size_t)(n0 + nn) * K + k0 + kk];
    }
  };
  auto stage = [&](int buf) {
#pragma unroll
    for (int i = 0; i < 16; ++i) {
      int idx = tid + i * 256;
      sA[buf][idx >> 5][idx & 31] = (bh)ga[i];
    }
#pragma unroll
    for (int i = 0; i < 2; ++i) {
      int v = tid + i * 256;
      *(uint4*)&sW[buf][v >> 2][(v & 3) << 3] = rw[i];
    }
  };

  v8f acc[2][4] = {};
  preload(0);
  int buf = 0;
  for (int k0 = 0; k0 < K; k0 += BK) {
    stage(buf);
    __syncthreads();
    if (k0 + BK < K) preload(k0 + BK);

    v16bf af[2], bf[4];
#pragma unroll
    for (int i = 0; i < 2; ++i)
      af[i] = frag_from_lds(&sA[buf][wm + i * 16 + r][0], kb);
#pragma unroll
    for (int j = 0; j < 4; ++j)
      bf[j] = frag_from_lds(&sW[buf][wn + j * 16 + r][0], kb);
#pragma unroll
    for (int i = 0; i < 2; ++i)
#pragma unroll
      for (int j = 0; j < 4; ++j)
        acc[i][j] = __builtin_amdgcn_wmma_f32_16x16x32_bf16(
            false, af[i], false, bf[j], (short)0, acc[i][j], false, false);
    buf ^= 1;
  }

  const int rhalf = (lane >> 4) << 3;
  for (int ti = 0; ti < 2; ++ti)
    for (int tj = 0; tj < 4; ++tj)
#pragma unroll
      for (int i = 0; i < 8; ++i) {
        int m = m0 + wm + ti * 16 + i + rhalf;
        int n = n0 + wn + tj * 16 + (lane & 15);
        float v = fmaxf(acc[ti][tj][i] + bias[n], 0.f);
        out[(size_t)n * M + m] = v;   // channel-major [OC][D^3]
      }
}

// ---------------------------------------------------------------------------
// Farthest point sampling: one block per batch, blockDim == n (power of 2).
// ---------------------------------------------------------------------------
__global__ void fps_kernel(const float* __restrict__ xyz, int n, int npoint,
                           int* __restrict__ idx_out, float* __restrict__ nxyz) {
  const int b = blockIdx.x, t = threadIdx.x;
  xyz     += (size_t)b * n * 3;
  idx_out += b * npoint;
  nxyz    += (size_t)b * npoint * 3;
  __shared__ float rv[1024];
  __shared__ int   ri[1024];
  __shared__ int   sel[256];
  __shared__ float last[3];
  float px = xyz[t * 3], py = xyz[t * 3 + 1], pz = xyz[t * 3 + 2];
  float dist = 1e10f;
  if (t == 0) { sel[0] = 0; last[0] = xyz[0]; last[1] = xyz[1]; last[2] = xyz[2]; }
  __syncthreads();
  for (int s = 1; s < npoint; ++s) {
    float dx = px - last[0], dy = py - last[1], dz = pz - last[2];
    dist = fminf(dist, dx * dx + dy * dy + dz * dz);
    rv[t] = dist; ri[t] = t;
    __syncthreads();
    for (int off = n >> 1; off > 0; off >>= 1) {
      if (t < off) {
        float v2 = rv[t + off]; int i2 = ri[t + off];
        if (v2 > rv[t] || (v2 == rv[t] && i2 < ri[t])) { rv[t] = v2; ri[t] = i2; }
      }
      __syncthreads();
    }
    if (t == 0) {
      int j = ri[0]; sel[s] = j;
      last[0] = xyz[j * 3]; last[1] = xyz[j * 3 + 1]; last[2] = xyz[j * 3 + 2];
    }
    __syncthreads();
  }
  if (t < npoint) {
    int j = sel[t];
    idx_out[t] = j;
    nxyz[t * 3 + 0] = xyz[j * 3 + 0];
    nxyz[t * 3 + 1] = xyz[j * 3 + 1];
    nxyz[t * 3 + 2] = xyz[j * 3 + 2];
  }
}

// ---------------------------------------------------------------------------
// SA0: ball(r=0.02, 64 samples from 1024) -> mlp 3->64->128 -> max.
// grid (256 centers, B), block 64.
// ---------------------------------------------------------------------------
__global__ __launch_bounds__(64)
void sa0_kernel(const float* __restrict__ xyz, const float* __restrict__ nxyz,
                const float* __restrict__ w0, const float* __restrict__ b0,
                const float* __restrict__ w1, const float* __restrict__ b1,
                float* __restrict__ fout /* [B][128][256] */) {
  const int c = blockIdx.x, b = blockIdx.y, t = threadIdx.x;
  xyz += (size_t)b * kNObj * 3;
  const float cx = nxyz[((size_t)b * 256 + c) * 3 + 0];
  const float cy = nxyz[((size_t)b * 256 + c) * 3 + 1];
  const float cz = nxyz[((size_t)b * 256 + c) * 3 + 2];
  __shared__ int   nbr[64];
  __shared__ float h0[64][64];  // [channel][sample]
  if (t == 0) {
    int cnt = 0;
    for (int j = 0; j < kNObj && cnt < 64; ++j) {
      float dx = xyz[j * 3] - cx, dy = xyz[j * 3 + 1] - cy, dz = xyz[j * 3 + 2] - cz;
      if (dx * dx + dy * dy + dz * dz < 0.02f * 0.02f) nbr[cnt++] = j;
    }
    int fill = (cnt > 0) ? nbr[0] : 0;
    for (int j2 = cnt; j2 < 64; ++j2) nbr[j2] = fill;
  }
  __syncthreads();
  int j = nbr[t];
  float gx = xyz[j * 3] - cx, gy = xyz[j * 3 + 1] - cy, gz = xyz[j * 3 + 2] - cz;
  for (int oc = 0; oc < 64; ++oc) {
    float v = w0[oc * 3] * gx + w0[oc * 3 + 1] * gy + w0[oc * 3 + 2] * gz + b0[oc];
    h0[oc][t] = fmaxf(v, 0.f);
  }
  __syncthreads();
  for (int o = t; o < 128; o += 64) {
    float mx = 0.f;  // relu outputs are >= 0
    for (int s = 0; s < 64; ++s) {
      float a = b1[o];
      for (int cc = 0; cc < 64; ++cc) a += w1[o * 64 + cc] * h0[cc][s];
      mx = fmaxf(mx, fmaxf(a, 0.f));
    }
    fout[((size_t)b * 128 + o) * 256 + c] = mx;
  }
}

// ---------------------------------------------------------------------------
// SA1: ball(r=0.04, 128 samples from 256 pts, feats 128) -> 131->128->256 -> max.
// grid (64 centers, B), block 128; samples processed in 2 chunks of 64.
// ---------------------------------------------------------------------------
__global__ __launch_bounds__(128)
void sa1_kernel(const float* __restrict__ nx0, const float* __restrict__ f0,
                const float* __restrict__ nx1,
                const float* __restrict__ w0, const float* __restrict__ b0,
                const float* __restrict__ w1, const float* __restrict__ b1,
                float* __restrict__ fout /* [B][256][64] */) {
  const int c = blockIdx.x, b = blockIdx.y, t = threadIdx.x;
  const float* pts = nx0 + (size_t)b * 256 * 3;
  const float* fb  = f0 + (size_t)b * 128 * 256;
  const float cx = nx1[((size_t)b * 64 + c) * 3 + 0];
  const float cy = nx1[((size_t)b * 64 + c) * 3 + 1];
  const float cz = nx1[((size_t)b * 64 + c) * 3 + 2];
  __shared__ int   nbr[128];
  __shared__ float hs[128][64];  // [channel][sample-in-chunk]  32 KB
  if (t == 0) {
    int cnt = 0;
    for (int j = 0; j < 256 && cnt < 128; ++j) {
      float dx = pts[j * 3] - cx, dy = pts[j * 3 + 1] - cy, dz = pts[j * 3 + 2] - cz;
      if (dx * dx + dy * dy + dz * dz < 0.04f * 0.04f) nbr[cnt++] = j;
    }
    int fill = (cnt > 0) ? nbr[0] : 0;
    for (int j2 = cnt; j2 < 128; ++j2) nbr[j2] = fill;
  }
  __syncthreads();
  float mx[2] = {0.f, 0.f};
  for (int ch = 0; ch < 2; ++ch) {
    int sl = t & 63, half = t >> 6;
    int j = nbr[ch * 64 + sl];
    float gx = pts[j * 3] - cx, gy = pts[j * 3 + 1] - cy, gz = pts[j * 3 + 2] - cz;
    for (int oc = half * 64; oc < half * 64 + 64; ++oc) {
      const float* wr = w0 + (size_t)oc * 131;
      float a = wr[0] * gx + wr[1] * gy + wr[2] * gz + b0[oc];
      for (int cc = 0; cc < 128; ++cc) a += wr[3 + cc] * fb[cc * 256 + j];
      hs[oc][sl] = fmaxf(a, 0.f);
    }
    __syncthreads();
    for (int oi = 0; oi < 2; ++oi) {
      int o = t + oi * 128;
      float m = mx[oi];
      for (int s2 = 0; s2 < 64; ++s2) {
        float a = b1[o];
        for (int cc = 0; cc < 128; ++cc) a += w1[o * 128 + cc] * hs[cc][s2];
        m = fmaxf(m, fmaxf(a, 0.f));
      }
      mx[oi] = m;
    }
    __syncthreads();
  }
  fout[((size_t)b * 256 + t) * 64 + c]       = mx[0];
  fout[((size_t)b * 256 + t + 128) * 64 + c] = mx[1];
}

// ---------------------------------------------------------------------------
// SA2: 259 -> 256 -> 512, max over 64 points. grid (B), block 256.
// ---------------------------------------------------------------------------
__global__ __launch_bounds__(256)
void sa2_kernel(const float* __restrict__ nx1, const float* __restrict__ f1,
                const float* __restrict__ w0, const float* __restrict__ b0,
                const float* __restrict__ w1, const float* __restrict__ b1,
                float* __restrict__ f2 /* [B][512] */) {
  const int b = blockIdx.x, t = threadIdx.x;
  const float* px = nx1 + (size_t)b * 64 * 3;
  const float* fb = f1 + (size_t)b * 256 * 64;
  __shared__ float hs[256][32];  // 32 KB
  float mx[2] = {0.f, 0.f};
  for (int ch = 0; ch < 2; ++ch) {
    const float* wr = w0 + (size_t)t * 259;
    for (int p = 0; p < 32; ++p) {
      int pt = ch * 32 + p;
      float a = wr[0] * px[pt * 3] + wr[1] * px[pt * 3 + 1] +
                wr[2] * px[pt * 3 + 2] + b0[t];
      for (int cc = 0; cc < 256; ++cc) a += wr[3 + cc] * fb[cc * 64 + pt];
      hs[t][p] = fmaxf(a, 0.f);
    }
    __syncthreads();
    for (int oi = 0; oi < 2; ++oi) {
      int o = t + oi * 256;
      float m = mx[oi];
      for (int p = 0; p < 32; ++p) {
        float a = b1[o];
        for (int cc = 0; cc < 256; ++cc) a += w1[(size_t)o * 256 + cc] * hs[cc][p];
        m = fmaxf(m, fmaxf(a, 0.f));
      }
      mx[oi] = m;
    }
    __syncthreads();
  }
  f2[(size_t)b * 512 + t]       = mx[0];
  f2[(size_t)b * 512 + t + 256] = mx[1];
}

// Small per-batch FC + relu: out[b][n] = relu(W[n,:] . in[b,:] + bias[n])
__global__ void fc_relu_kernel(const float* __restrict__ in,
                               const float* __restrict__ W,
                               const float* __restrict__ bias,
                               float* __restrict__ out, int K, int N) {
  int b = blockIdx.y;
  int n = blockIdx.x * blockDim.x + threadIdx.x;
  if (n >= N) return;
  const float* x  = in + (size_t)b * K;
  const float* wr = W + (size_t)n * K;
  float a = bias[n];
  for (int k = 0; k < K; ++k) a += wr[k] * x[k];
  out[(size_t)b * N + n] = fmaxf(a, 0.f);
}

// ---------------------------------------------------------------------------
// Scene PointNet fused with voxel scatter-max. One block per point (128 thr).
// vox layout [B][256][4096]; atomicMax on int bits (values >= 0).
// ---------------------------------------------------------------------------
__global__ __launch_bounds__(128)
void pointnet_kernel(const float* __restrict__ scene,
                     const float* __restrict__ w0, const float* __restrict__ b0,
                     const float* __restrict__ w1, const float* __restrict__ b1,
                     float* __restrict__ vox) {
  const int p = blockIdx.x, b = blockIdx.y, t = threadIdx.x;
  const float* xp = scene + ((size_t)b * kNScene + p) * 3;
  float x = xp[0], y = xp[1], z = xp[2];
  int ix = min(15, max(0, (int)floorf((x - kLow) / kVox)));
  int iy = min(15, max(0, (int)floorf((y - kLow) / kVox)));
  int iz = min(15, max(0, (int)floorf((z - kLow) / kVox)));
  int flat = ix * 256 + iy * 16 + iz;
  float ox = x - (ix * kVox + kVox * 0.5f + kLow);
  float oy = y - (iy * kVox + kVox * 0.5f + kLow);
  float oz = z - (iz * kVox + kVox * 0.5f + kLow);
  __shared__ float h[128];
  float a = w0[t * 3] * ox + w0[t * 3 + 1] * oy + w0[t * 3 + 2] * oz + b0[t];
  h[t] = fmaxf(a, 0.f);
  __syncthreads();
  int* vb = (int*)(vox + (size_t)b * 256 * 4096);
  for (int oi = 0; oi < 2; ++oi) {
    int o = t + oi * 128;
    const float* wr = w1 + (size_t)o * 128;
    float acc = b1[o];
    for (int cc = 0; cc < 128; ++cc) acc += wr[cc] * h[cc];
    acc = fmaxf(acc, 0.f);
    atomicMax(vb + (size_t)o * 4096 + flat, __float_as_int(acc));
  }
}

// Max-pool 2x2x2 stride 2, channel-major [BC][D^3] -> [BC][(D/2)^3].
__global__ void maxpool2_kernel(const float* __restrict__ in,
                                float* __restrict__ out, int D, int total) {
  int i = blockIdx.x * 256 + threadIdx.x;
  if (i >= total) return;
  int Dh = D >> 1, vol = Dh * Dh * Dh;
  int v = i % vol, bc = i / vol;
  int z = v % Dh, y = (v / Dh) % Dh, x = v / (Dh * Dh);
  const float* src = in + (size_t)bc * D * D * D;
  float m = -3.402823466e38f;
  for (int dx = 0; dx < 2; ++dx)
    for (int dy = 0; dy < 2; ++dy)
      for (int dz = 0; dz < 2; ++dz)
        m = fmaxf(m, src[((2 * x + dx) * D + (2 * y + dy)) * D + (2 * z + dz)]);
  out[i] = m;
}

// Transposed conv 2x2x2 stride 2: v2 [B][1024][4^3] -> v3 [B][512][8^3].
__global__ void convt_kernel(const float* __restrict__ v2,
                             const float* __restrict__ wt,
                             const float* __restrict__ bt,
                             float* __restrict__ v3) {
  int i = blockIdx.x * 256 + threadIdx.x;
  if (i >= kB * 512 * 512) return;
  int v = i % 512, o = (i / 512) % 512, b = i / (512 * 512);
  int gx = v >> 6, gy = (v >> 3) & 7, gz = v & 7;
  int ii = gx >> 1, jj = gy >> 1, kk = gz >> 1;
  int xx = gx & 1, yy = gy & 1, zz = gz & 1;
  const float* src = v2 + (size_t)b * 1024 * 64 + (ii * 4 + jj) * 4 + kk;
  const float* w   = wt + (size_t)o * 8 + (xx * 4 + yy * 2 + zz);
  float a = bt[o];
  for (int c = 0; c < 1024; ++c)
    a += src[(size_t)c * 64] * w[(size_t)c * 512 * 8];
  v3[i] = a;
}

// Build padded classifier input rows directly in bf16.
// cin row r=b*Q+q: [obj_f(1024) | scene_f(1024) | offs(3) | rots(6) | pad->2080]
__global__ __launch_bounds__(256)
void cin_build_kernel(const float* __restrict__ obj_f,
                      const float* __restrict__ v1p, const float* __restrict__ v3,
                      const float* __restrict__ trans, const float* __restrict__ rots,
                      bh* __restrict__ cin) {
  const int row = blockIdx.x;
  const int b = row / kQ, q = row % kQ;
  const int t = threadIdx.x;
  float tx = trans[q * 3], ty = trans[q * 3 + 1], tz = trans[q * 3 + 2];
  int ix = min(7, max(0, (int)floorf((tx - kLow) / (2.f * kVox))));
  int iy = min(7, max(0, (int)floorf((ty - kLow) / (2.f * kVox))));
  int iz = min(7, max(0, (int)floorf((tz - kLow) / (2.f * kVox))));
  int flat = ix * 64 + iy * 8 + iz;
  float ox = tx - (ix * 0.125f + 0.03125f + kLow);
  float oy = ty - (iy * 0.125f + 0.03125f + kLow);
  float oz = tz - (iz * 0.125f + 0.03125f + kLow);
  bh* dst = cin + (size_t)row * 2080;
  for (int col = t; col < 2080; col += 256) {
    float v;
    if (col < 1024)       v = obj_f[b * 1024 + col];
    else if (col < 1536)  v = v1p[((size_t)b * 512 + (col - 1024)) * 512 + flat];
    else if (col < 2048)  v = v3[((size_t)b * 512 + (col - 1536)) * 512 + flat];
    else if (col == 2048) v = ox;
    else if (col == 2049) v = oy;
    else if (col == 2050) v = oz;
    else if (col < 2057)  v = rots[q * 6 + (col - 2051)];
    else                  v = 0.f;
    dst[col] = (bh)v;
  }
}

// fp32 -> bf16 with optional K-padding (dst row stride ldd >= K, pad zeros).
__global__ void cvt_bf16_kernel(const float* __restrict__ src,
                                bh* __restrict__ dst, int K, int ldd,
                                long long total) {
  long long i = (long long)blockIdx.x * 256 + threadIdx.x;
  if (i >= total) return;
  long long r = i / ldd;
  int c = (int)(i - r * ldd);
  dst[i] = (c < K) ? (bh)src[r * K + c] : (bh)0.f;
}

// Final 256 -> 1 layer.
__global__ void cls2_kernel(const float* __restrict__ h2,
                            const float* __restrict__ w,
                            const float* __restrict__ bias,
                            float* __restrict__ out) {
  int r = blockIdx.x * 256 + threadIdx.x;
  if (r >= kB * kQ) return;
  const float* x = h2 + (size_t)r * 256;
  float a = bias[0];
  for (int k = 0; k < 256; ++k) a += x[k] * w[k];
  out[r] = a;
}

// ---------------------------------------------------------------------------
// Orchestration
// ---------------------------------------------------------------------------
extern "C" void kernel_launch(void* const* d_in, const int* in_sizes, int n_in,
                              void* d_out, int out_size, void* d_ws, size_t ws_size,
                              hipStream_t stream) {
  (void)in_sizes; (void)n_in; (void)out_size; (void)ws_size;
  const float* scene = (const float*)d_in[0];
  const float* objpc = (const float*)d_in[1];
  const float* trans = (const float*)d_in[2];
  const float* rots  = (const float*)d_in[3];
  int pi = 4;
  const float* sa0_w0 = (const float*)d_in[pi++];
  const float* sa0_b0 = (const float*)d_in[pi++];
  const float* sa0_w1 = (const float*)d_in[pi++];
  const float* sa0_b1 = (const float*)d_in[pi++];
  const float* sa1_w0 = (const float*)d_in[pi++];
  const float* sa1_b0 = (const float*)d_in[pi++];
  const float* sa1_w1 = (const float*)d_in[pi++];
  const float* sa1_b1 = (const float*)d_in[pi++];
  const float* sa2_w0 = (const float*)d_in[pi++];
  const float* sa2_b0 = (const float*)d_in[pi++];
  const float* sa2_w1 = (const float*)d_in[pi++];
  const float* sa2_b1 = (const float*)d_in[pi++];
  const float* ofc0_w = (const float*)d_in[pi++];
  const float* ofc0_b = (const float*)d_in[pi++];
  const float* ofc1_w = (const float*)d_in[pi++];
  const float* ofc1_b = (const float*)d_in[pi++];
  const float* pt0_w  = (const float*)d_in[pi++];
  const float* pt0_b  = (const float*)d_in[pi++];
  const float* pt1_w  = (const float*)d_in[pi++];
  const float* pt1_b  = (const float*)d_in[pi++];
  const float* c0_w   = (const float*)d_in[pi++];
  const float* c0_b   = (const float*)d_in[pi++];
  const float* c1_w   = (const float*)d_in[pi++];
  const float* c1_b   = (const float*)d_in[pi++];
  const float* ct_w   = (const float*)d_in[pi++];
  const float* ct_b   = (const float*)d_in[pi++];
  const float* cls0_w = (const float*)d_in[pi++];
  const float* cls0_b = (const float*)d_in[pi++];
  const float* cls1_w = (const float*)d_in[pi++];
  const float* cls1_b = (const float*)d_in[pi++];
  const float* cls2_w = (const float*)d_in[pi++];
  const float* cls2_b = (const float*)d_in[pi++];

  // Workspace carve (256B aligned).
  char* wsp = (char*)d_ws;
  auto alloc = [&](size_t bytes) -> void* {
    void* p = wsp;
    wsp += (bytes + 255) & ~(size_t)255;
    return p;
  };
  int*   idx0  = (int*)  alloc(kB * 256 * sizeof(int));
  float* nx0   = (float*)alloc((size_t)kB * 256 * 3 * 4);
  float* f0    = (float*)alloc((size_t)kB * 128 * 256 * 4);
  int*   idx1  = (int*)  alloc(kB * 64 * sizeof(int));
  float* nx1   = (float*)alloc((size_t)kB * 64 * 3 * 4);
  float* f1    = (float*)alloc((size_t)kB * 256 * 64 * 4);
  float* f2    = (float*)alloc((size_t)kB * 512 * 4);
  float* objh  = (float*)alloc((size_t)kB * 1024 * 4);
  float* objf  = (float*)alloc((size_t)kB * 1024 * 4);
  float* vox   = (float*)alloc((size_t)kB * 256 * 4096 * 4);
  float* v1    = (float*)alloc((size_t)kB * 512 * 4096 * 4);
  float* v1p   = (float*)alloc((size_t)kB * 512 * 512 * 4);
  float* u2    = (float*)alloc((size_t)kB * 1024 * 512 * 4);
  float* v2p   = (float*)alloc((size_t)kB * 1024 * 64 * 4);
  float* v3    = (float*)alloc((size_t)kB * 512 * 512 * 4);
  bh*    wc0   = (bh*)   alloc((size_t)512 * 6912 * 2);
  bh*    wc1   = (bh*)   alloc((size_t)1024 * 13824 * 2);
  bh*    wcls0 = (bh*)   alloc((size_t)1024 * 2080 * 2);
  bh*    wcls1 = (bh*)   alloc((size_t)256 * 1024 * 2);
  bh*    cin   = (bh*)   alloc((size_t)kB * kQ * 2080 * 2);
  float* h1    = (float*)alloc((size_t)kB * kQ * 1024 * 4);
  bh*    h1b   = (bh*)   alloc((size_t)kB * kQ * 1024 * 2);
  float* h2    = (float*)alloc((size_t)kB * kQ * 256 * 4);

  // --- object encoder (scalar; negligible FLOPs) ---
  fps_kernel<<<kB, 1024, 0, stream>>>(objpc, 1024, 256, idx0, nx0);
  sa0_kernel<<<dim3(256, kB), 64, 0, stream>>>(objpc, nx0, sa0_w0, sa0_b0, sa0_w1, sa0_b1, f0);
  fps_kernel<<<kB, 256, 0, stream>>>(nx0, 256, 64, idx1, nx1);
  sa1_kernel<<<dim3(64, kB), 128, 0, stream>>>(nx0, f0, nx1, sa1_w0, sa1_b0, sa1_w1, sa1_b1, f1);
  sa2_kernel<<<kB, 256, 0, stream>>>(nx1, f1, sa2_w0, sa2_b0, sa2_w1, sa2_b1, f2);
  fc_relu_kernel<<<dim3(4, kB), 256, 0, stream>>>(f2, ofc0_w, ofc0_b, objh, 512, 1024);
  fc_relu_kernel<<<dim3(4, kB), 256, 0, stream>>>(objh, ofc1_w, ofc1_b, objf, 1024, 1024);

  // --- scene encoder ---
  hipMemsetAsync(vox, 0, (size_t)kB * 256 * 4096 * 4, stream);
  pointnet_kernel<<<dim3(kNScene, kB), 128, 0, stream>>>(scene, pt0_w, pt0_b, pt1_w, pt1_b, vox);

  {
    long long t0 = (long long)512 * 6912;
    cvt_bf16_kernel<<<(int)((t0 + 255) / 256), 256, 0, stream>>>(c0_w, wc0, 6912, 6912, t0);
  }
  conv3d_wmma<16><<<dim3(32, 4, kB), 256, 0, stream>>>(vox, wc0, c0_b, v1, 256, 512);
  maxpool2_kernel<<<(kB * 512 * 512 + 255) / 256, 256, 0, stream>>>(v1, v1p, 16, kB * 512 * 512);

  {
    long long t1 = (long long)1024 * 13824;
    cvt_bf16_kernel<<<(int)((t1 + 255) / 256), 256, 0, stream>>>(c1_w, wc1, 13824, 13824, t1);
  }
  conv3d_wmma<8><<<dim3(4, 8, kB), 256, 0, stream>>>(v1p, wc1, c1_b, u2, 512, 1024);
  maxpool2_kernel<<<(kB * 1024 * 64 + 255) / 256, 256, 0, stream>>>(u2, v2p, 8, kB * 1024 * 64);
  convt_kernel<<<(kB * 512 * 512 + 255) / 256, 256, 0, stream>>>(v2p, ct_w, ct_b, v3);

  // --- classifier (WMMA GEMMs) ---
  {
    long long t2 = (long long)1024 * 2080;
    cvt_bf16_kernel<<<(int)((t2 + 255) / 256), 256, 0, stream>>>(cls0_w, wcls0, 2057, 2080, t2);
    long long t3 = (long long)256 * 1024;
    cvt_bf16_kernel<<<(int)((t3 + 255) / 256), 256, 0, stream>>>(cls1_w, wcls1, 1024, 1024, t3);
  }
  cin_build_kernel<<<kB * kQ, 256, 0, stream>>>(objf, v1p, v3, trans, rots, cin);
  gemm_bf16_relu<<<dim3(32, 8), 256, 0, stream>>>(cin, wcls0, cls0_b, h1, kB * kQ, 1024, 2080, 1);
  {
    long long t4 = (long long)kB * kQ * 1024;
    cvt_bf16_kernel<<<(int)((t4 + 255) / 256), 256, 0, stream>>>(h1, h1b, 1024, 1024, t4);
  }
  gemm_bf16_relu<<<dim3(32, 2), 256, 0, stream>>>(h1b, wcls1, cls1_b, h2, kB * kQ, 256, 1024, 1);
  cls2_kernel<<<(kB * kQ + 255) / 256, 256, 0, stream>>>(h2, cls2_w, cls2_b, (float*)d_out);
}
